// TextGeneratorModel_56564719289056
// MI455X (gfx1250) — matrix operand order repaired and checked by reading, not compiled, not run on hardware
//
#include <hip/hip_runtime.h>
#include <hip/hip_bf16.h>

// Problem constants (from reference)
#define B_ 8
#define T_ 256
#define V_ 50257
#define VP_ 50432          // 197*256: padded Wd leading dim for bf16 fast path
#define E_ 256
#define H1_ 512
#define H2_ 1024
#define MROWS_ (B_ * T_)   // 2048 GEMM rows, row index = b*T + t

typedef __attribute__((ext_vector_type(16))) __bf16 v16bf;
typedef __attribute__((ext_vector_type(8)))  float  v8f;

__device__ __forceinline__ unsigned short f2bf(float f) {
  // round-to-nearest-even f32 -> bf16
  unsigned u = __float_as_uint(f);
  u += 0x7FFFu + ((u >> 16) & 1u);
  return (unsigned short)(u >> 16);
}

__device__ __forceinline__ float sigf(float x) {
  return 1.0f / (1.0f + __expf(-x));
}

__device__ __forceinline__ v8f wmma_bf16(v16bf a, v16bf b, v8f c) {
  // (neg_a, A, neg_b, B, c_mod, C, reuse_a, reuse_b)
  return __builtin_amdgcn_wmma_f32_16x16x32_bf16(false, a, false, b, (short)0, c,
                                                 false, false);
}

union FragU {
  v16bf v;
  uint4 u[2];
  unsigned short s[16];
};

// A fragment (16x32 bf16, row-major source). ISA layout:
//   lane L: m = m0 + (L&15); element e holds k = k0 + 16*(e>>3) + 8*(L>>4) + (e&7)
__device__ __forceinline__ v16bf load_a_frag(const unsigned short* A, int lda,
                                             int m0, int k0, int lane) {
  int half = (lane >> 4) & 1;
  const unsigned short* p =
      A + (long)(m0 + (lane & 15)) * lda + k0 + 8 * half;
  FragU t;
  t.u[0] = *(const uint4*)(p);        // k0+8h .. k0+8h+7
  t.u[1] = *(const uint4*)(p + 16);   // k0+16+8h .. +7
  return t.v;
}

// B fragment (32x16 bf16, row-major source). lane L supplies column n0+(L&15),
// element e holds k = k0 + 16*(e>>3) + 8*(L>>4) + (e&7).
__device__ __forceinline__ v16bf load_b_frag(const unsigned short* Bm, long ldb,
                                             int k0, int n0, int lane) {
  const unsigned short* p =
      Bm + (long)(k0 + 8 * ((lane >> 4) & 1)) * ldb + n0 + (lane & 15);
  FragU t;
#pragma unroll
  for (int e = 0; e < 16; ++e)
    t.s[e] = p[(long)((e & 7) + 16 * (e >> 3)) * ldb];
  return t.v;
}

// B fragment from f32 source, converted to bf16 in registers, N-guarded.
__device__ __forceinline__ v16bf load_b_frag_f32(const float* Bm, long ldb,
                                                 int k0, int n0, int nmax,
                                                 int lane) {
  const float* p =
      Bm + (long)(k0 + 8 * ((lane >> 4) & 1)) * ldb + n0 + (lane & 15);
  bool ok = (n0 + (lane & 15)) < nmax;
  FragU t;
#pragma unroll
  for (int e = 0; e < 16; ++e) {
    float f = ok ? p[(long)((e & 7) + 16 * (e >> 3)) * ldb] : 0.0f;
    t.s[e] = f2bf(f);
  }
  return t.v;
}

// ---------------------------------------------------------------------------
// Utility kernels
// ---------------------------------------------------------------------------
__global__ void zero_kernel(unsigned int* p, int nwords) {
  int i = blockIdx.x * blockDim.x + threadIdx.x;
  if (i < nwords) p[i] = 0u;
}

__global__ void cvt_bf16_kernel(const float* __restrict__ src,
                                unsigned short* __restrict__ dst, long n) {
  long i = (long)blockIdx.x * blockDim.x + threadIdx.x;
  if (i < n) dst[i] = f2bf(src[i]);
}

// Wd [K, V] f32 -> Wdbf [K, VP_] bf16, zero-filled pad columns.
__global__ void cvt_pad_bf16_kernel(const float* __restrict__ src,
                                    unsigned short* __restrict__ dst, long n) {
  long i = (long)blockIdx.x * blockDim.x + threadIdx.x;
  if (i >= n) return;
  long row = i / VP_;
  int col = (int)(i - row * VP_);
  dst[i] = (col < V_) ? f2bf(src[row * V_ + col]) : (unsigned short)0;
}

__global__ void gather_emb_kernel(const int* __restrict__ tokens,
                                  const float* __restrict__ emb,
                                  unsigned short* __restrict__ Xbf, int n) {
  int i = blockIdx.x * blockDim.x + threadIdx.x;
  if (i >= n) return;
  int row = i / E_;
  int e = i - row * E_;
  int tok = tokens[row];
  Xbf[i] = f2bf(emb[(long)tok * E_ + e]);
}

// ---------------------------------------------------------------------------
// Generic bf16 x bf16 -> f32 GEMM with bias:  C[M,N] = A[M,K] * B[K,N] + bias
// WG = 256 threads = 8 waves, block tile 64(M) x 256(N), wave tile 32x64.
// ---------------------------------------------------------------------------
__global__ __launch_bounds__(256) void gemm_bf16_kernel(
    const unsigned short* __restrict__ A, const unsigned short* __restrict__ Bm,
    const float* __restrict__ bias, float* __restrict__ C, int M, int N, int K) {
  int n0 = blockIdx.x * 256;
  int m0 = blockIdx.y * 64;
  int wave = threadIdx.x >> 5;
  int lane = threadIdx.x & 31;
  int mw = m0 + (wave >> 2) * 32;  // 2 wave rows
  int nw = n0 + (wave & 3) * 64;   // 4 wave cols

  v8f acc[2][4];
#pragma unroll
  for (int i = 0; i < 2; ++i)
#pragma unroll
    for (int j = 0; j < 4; ++j)
      acc[i][j] = (v8f){0.f, 0.f, 0.f, 0.f, 0.f, 0.f, 0.f, 0.f};

  for (int k0 = 0; k0 < K; k0 += 32) {
    v16bf a0 = load_a_frag(A, K, mw, k0, lane);
    v16bf a1 = load_a_frag(A, K, mw + 16, k0, lane);
#pragma unroll
    for (int j = 0; j < 4; ++j) {
      v16bf b = load_b_frag(Bm, N, k0, nw + j * 16, lane);
      acc[0][j] = wmma_bf16(a0, b, acc[0][j]);
      acc[1][j] = wmma_bf16(a1, b, acc[1][j]);
    }
  }

  int half = lane >> 4;
  int nl = lane & 15;
#pragma unroll
  for (int i = 0; i < 2; ++i)
#pragma unroll
    for (int j = 0; j < 4; ++j)
#pragma unroll
      for (int r = 0; r < 8; ++r) {
        int m = mw + i * 16 + r + 8 * half;
        int n = nw + j * 16 + nl;
        C[(long)m * N + n] = acc[i][j][r] + bias[n];
      }
}

// ---------------------------------------------------------------------------
// Fast final dense: logits[M,V] = Y2bf * Wdbf(bf16, padded ldb=VP_) + bd
// B reads always in-bounds thanks to padding; stores N-guarded, non-temporal.
// ---------------------------------------------------------------------------
__global__ __launch_bounds__(256) void dense_bf16_kernel(
    const unsigned short* __restrict__ A, const unsigned short* __restrict__ Bm,
    const float* __restrict__ bd, float* __restrict__ out, int M, int N, int K) {
  int n0 = blockIdx.x * 256;
  int m0 = blockIdx.y * 64;
  int wave = threadIdx.x >> 5;
  int lane = threadIdx.x & 31;
  int mw = m0 + (wave >> 2) * 32;
  int nw = n0 + (wave & 3) * 64;

  v8f acc[2][4];
#pragma unroll
  for (int i = 0; i < 2; ++i)
#pragma unroll
    for (int j = 0; j < 4; ++j)
      acc[i][j] = (v8f){0.f, 0.f, 0.f, 0.f, 0.f, 0.f, 0.f, 0.f};

  for (int k0 = 0; k0 < K; k0 += 32) {
    v16bf a0 = load_a_frag(A, K, mw, k0, lane);
    v16bf a1 = load_a_frag(A, K, mw + 16, k0, lane);
#pragma unroll
    for (int j = 0; j < 4; ++j) {
      v16bf b = load_b_frag(Bm, (long)VP_, k0, nw + j * 16, lane);
      acc[0][j] = wmma_bf16(a0, b, acc[0][j]);
      acc[1][j] = wmma_bf16(a1, b, acc[1][j]);
    }
  }

  int half = lane >> 4;
  int nl = lane & 15;
#pragma unroll
  for (int i = 0; i < 2; ++i)
#pragma unroll
    for (int j = 0; j < 4; ++j)
#pragma unroll
      for (int r = 0; r < 8; ++r) {
        int m = mw + i * 16 + r + 8 * half;
        int n = nw + j * 16 + nl;
        if (n < N)
          __builtin_nontemporal_store(acc[i][j][r] + bd[n],
                                      &out[(long)m * N + n]);
      }
}

// ---------------------------------------------------------------------------
// Fallback final dense (small workspace): Wd f32 converted in-register.
// ---------------------------------------------------------------------------
__global__ __launch_bounds__(256) void dense_out_kernel(
    const unsigned short* __restrict__ A, const float* __restrict__ Wd,
    const float* __restrict__ bd, float* __restrict__ out, int M, int N, int K) {
  int n0 = blockIdx.x * 256;
  int m0 = blockIdx.y * 64;
  int wave = threadIdx.x >> 5;
  int lane = threadIdx.x & 31;
  int mw = m0 + (wave >> 2) * 32;
  int nw = n0 + (wave & 3) * 64;

  v8f acc[2][4];
#pragma unroll
  for (int i = 0; i < 2; ++i)
#pragma unroll
    for (int j = 0; j < 4; ++j)
      acc[i][j] = (v8f){0.f, 0.f, 0.f, 0.f, 0.f, 0.f, 0.f, 0.f};

  for (int k0 = 0; k0 < K; k0 += 32) {
    v16bf a0 = load_a_frag(A, K, mw, k0, lane);
    v16bf a1 = load_a_frag(A, K, mw + 16, k0, lane);
#pragma unroll
    for (int j = 0; j < 4; ++j) {
      v16bf b = load_b_frag_f32(Wd, (long)N, k0, nw + j * 16, N, lane);
      acc[0][j] = wmma_bf16(a0, b, acc[0][j]);
      acc[1][j] = wmma_bf16(a1, b, acc[1][j]);
    }
  }

  int half = lane >> 4;
  int nl = lane & 15;
#pragma unroll
  for (int i = 0; i < 2; ++i)
#pragma unroll
    for (int j = 0; j < 4; ++j)
#pragma unroll
      for (int r = 0; r < 8; ++r) {
        int m = mw + i * 16 + r + 8 * half;
        int n = nw + j * 16 + nl;
        if (n < N)
          __builtin_nontemporal_store(acc[i][j][r] + bd[n],
                                      &out[(long)m * N + n]);
      }
}

// ---------------------------------------------------------------------------
// Masked LSTM recurrence (one layer). nWG workgroups, each owning 64 hidden
// columns (and the 4 matching gate slices). Per step:
//   z = xw[:, gatecols] + h(padded 16 rows, bf16) @ U[:, gatecols]  (WMMA)
//   gates in LDS -> elementwise sigmoid/tanh, Keras masking, h/c update
//   h_next written bf16 to double-buffered global h; per-step atomic barrier.
// ---------------------------------------------------------------------------
#define LCOLS 64
__global__ __launch_bounds__(256) void lstm_kernel(
    const float* __restrict__ xw,        // [B*T, 4H] (row = b*T + t)
    const unsigned short* __restrict__ U,  // [H, 4H] bf16
    unsigned short* __restrict__ hbuf,     // [2][16][H] bf16 (rows 8..15 zero)
    unsigned short* __restrict__ yout,     // [B*T, H] bf16
    const int* __restrict__ tokens,        // [B, T]
    int* __restrict__ ctr,                 // [T] arrive counters (pre-zeroed)
    int H, int nWG) {
  const int fourH = 4 * H;
  const int j0 = blockIdx.x * LCOLS;
  const int tid = threadIdx.x;
  const int wave = tid >> 5;
  const int lane = tid & 31;
  const int gate = wave >> 1;     // 0:i 1:f 2:g 3:o
  const int nhalf = wave & 1;     // which 32-col half of this WG's slice
  const int half = lane >> 4;
  const int nl = lane & 15;

  __shared__ float zs[4][B_][LCOLS];  // pre-activation gates
  __shared__ float cs[B_][LCOLS];     // cell state
  __shared__ float hp[B_][LCOLS];     // previous (post-mask) h

  for (int i = tid; i < B_ * LCOLS; i += 256) {
    (&cs[0][0])[i] = 0.f;
    (&hp[0][0])[i] = 0.f;
  }
  __syncthreads();

  for (int t = 0; t < T_; ++t) {
    const unsigned short* hcur = hbuf + (t & 1) * 16 * H;
    unsigned short* hnext = hbuf + ((t + 1) & 1) * 16 * H;

    // --- z-slice GEMM: 2 N-tiles per wave ---
    int nb0 = gate * H + j0 + nhalf * 32;
    v8f acc0 = (v8f){0.f, 0.f, 0.f, 0.f, 0.f, 0.f, 0.f, 0.f};
    v8f acc1 = acc0;
    for (int k0 = 0; k0 < H; k0 += 32) {
      v16bf a = load_a_frag(hcur, H, 0, k0, lane);
      v16bf b0 = load_b_frag(U, fourH, k0, nb0, lane);
      v16bf b1 = load_b_frag(U, fourH, k0, nb0 + 16, lane);
      acc0 = wmma_bf16(a, b0, acc0);
      acc1 = wmma_bf16(a, b1, acc1);
    }

    // --- add xw, stash z into LDS (only valid batch rows m<8) ---
#pragma unroll
    for (int j = 0; j < 2; ++j) {
      const v8f& a = j ? acc1 : acc0;
      int n = nb0 + j * 16 + nl;
      int jl = nhalf * 32 + j * 16 + nl;
#pragma unroll
      for (int r = 0; r < 8; ++r) {
        int m = r + 8 * half;
        if (m < B_)
          zs[gate][m][jl] = a[r] + xw[(long)(m * T_ + t) * fourH + n];
      }
    }
    __syncthreads();

    // --- elementwise gates + mask: 8*64 = 512 elements over 256 threads ---
#pragma unroll
    for (int rep = 0; rep < 2; ++rep) {
      int idx = tid + rep * 256;
      int b = idx >> 6;
      int jl = idx & 63;
      float iv = sigf(zs[0][b][jl]);
      float fv = sigf(zs[1][b][jl]);
      float gv = tanhf(zs[2][b][jl]);
      float ov = sigf(zs[3][b][jl]);
      float cn = fv * cs[b][jl] + iv * gv;
      float hn = ov * tanhf(cn);
      bool msk = tokens[b * T_ + t] != 0;
      float h2 = msk ? hn : hp[b][jl];
      float c2 = msk ? cn : cs[b][jl];
      cs[b][jl] = c2;
      hp[b][jl] = h2;
      unsigned short hb = f2bf(h2);
      hnext[b * H + j0 + jl] = hb;
      yout[(long)(b * T_ + t) * H + j0 + jl] = hb;
    }
    __threadfence();
    __syncthreads();

    // --- cross-workgroup step barrier (fresh counter per step) ---
    if (tid == 0) {
      __hip_atomic_fetch_add(&ctr[t], 1, __ATOMIC_RELEASE,
                             __HIP_MEMORY_SCOPE_AGENT);
      while (__hip_atomic_load(&ctr[t], __ATOMIC_ACQUIRE,
                               __HIP_MEMORY_SCOPE_AGENT) < nWG) {
        __builtin_amdgcn_s_sleep(2);
      }
    }
    __syncthreads();
    __threadfence();  // acquire visibility of other WGs' h writes
  }
}

// ---------------------------------------------------------------------------
extern "C" void kernel_launch(void* const* d_in, const int* in_sizes, int n_in,
                              void* d_out, int out_size, void* d_ws,
                              size_t ws_size, hipStream_t stream) {
  const int* tokens = (const int*)d_in[0];
  const float* emb = (const float*)d_in[1];
  const float* W1 = (const float*)d_in[2];
  const float* U1 = (const float*)d_in[3];
  const float* b1 = (const float*)d_in[4];
  const float* W2 = (const float*)d_in[5];
  const float* U2 = (const float*)d_in[6];
  const float* b2 = (const float*)d_in[7];
  const float* Wd = (const float*)d_in[8];
  const float* bd = (const float*)d_in[9];
  float* out = (float*)d_out;

  char* w = (char*)d_ws;
  size_t off = 0;
  auto alloc = [&](size_t bytes) -> void* {
    void* p = w + off;
    off += (bytes + 255) & ~(size_t)255;
    return p;
  };

  // Zero-initialized region first (counters + padded h double buffers)
  int* ctr = (int*)alloc(2 * T_ * sizeof(int));               // [2][T]
  unsigned short* h1buf = (unsigned short*)alloc(2 * 16 * H1_ * 2);
  unsigned short* h2buf = (unsigned short*)alloc(2 * 16 * H2_ * 2);
  size_t zeroWords = off / 4;

  unsigned short* Xbf  = (unsigned short*)alloc((size_t)MROWS_ * E_ * 2);
  unsigned short* W1bf = (unsigned short*)alloc((size_t)E_ * 4 * H1_ * 2);
  unsigned short* U1bf = (unsigned short*)alloc((size_t)H1_ * 4 * H1_ * 2);
  unsigned short* W2bf = (unsigned short*)alloc((size_t)H1_ * 4 * H2_ * 2);
  unsigned short* U2bf = (unsigned short*)alloc((size_t)H2_ * 4 * H2_ * 2);
  unsigned short* Y1bf = (unsigned short*)alloc((size_t)MROWS_ * H1_ * 2);
  unsigned short* Y2bf = (unsigned short*)alloc((size_t)MROWS_ * H2_ * 2);
  float* xw1 = (float*)alloc((size_t)MROWS_ * 4 * H1_ * 4);
  float* xw2 = (float*)alloc((size_t)MROWS_ * 4 * H2_ * 4);

  // Optional fast-path buffer: Wd converted to bf16 with padded stride.
  unsigned short* Wdbf = (unsigned short*)alloc((size_t)H2_ * VP_ * 2);
  bool fast_dense = (off <= ws_size);   // fixed per harness -> deterministic

  // 0) zero counters + h buffers (every launch -> graph-replay safe)
  zero_kernel<<<((int)zeroWords + 255) / 256, 256, 0, stream>>>(
      (unsigned int*)d_ws, (int)zeroWords);

  // 1) weight converts f32 -> bf16
  {
    long n;
    n = (long)E_ * 4 * H1_;
    cvt_bf16_kernel<<<(int)((n + 255) / 256), 256, 0, stream>>>(W1, W1bf, n);
    n = (long)H1_ * 4 * H1_;
    cvt_bf16_kernel<<<(int)((n + 255) / 256), 256, 0, stream>>>(U1, U1bf, n);
    n = (long)H1_ * 4 * H2_;
    cvt_bf16_kernel<<<(int)((n + 255) / 256), 256, 0, stream>>>(W2, W2bf, n);
    n = (long)H2_ * 4 * H2_;
    cvt_bf16_kernel<<<(int)((n + 255) / 256), 256, 0, stream>>>(U2, U2bf, n);
    if (fast_dense) {
      n = (long)H2_ * VP_;
      cvt_pad_bf16_kernel<<<(int)((n + 255) / 256), 256, 0, stream>>>(Wd, Wdbf,
                                                                      n);
    }
  }

  // 2) embedding gather -> bf16
  {
    int n = MROWS_ * E_;
    gather_emb_kernel<<<(n + 255) / 256, 256, 0, stream>>>(tokens, emb, Xbf, n);
  }

  // 3) xw1 = X @ W1 + b1   [2048 x 2048]
  gemm_bf16_kernel<<<dim3(4 * H1_ / 256, MROWS_ / 64), 256, 0, stream>>>(
      Xbf, W1bf, b1, xw1, MROWS_, 4 * H1_, E_);

  // 4) LSTM layer 1 (8 WGs x 64 cols), writes Y1bf
  lstm_kernel<<<H1_ / LCOLS, 256, 0, stream>>>(xw1, U1bf, h1buf, Y1bf, tokens,
                                               ctr, H1_, H1_ / LCOLS);

  // 5) xw2 = Y1 @ W2 + b2   [2048 x 4096]
  gemm_bf16_kernel<<<dim3(4 * H2_ / 256, MROWS_ / 64), 256, 0, stream>>>(
      Y1bf, W2bf, b2, xw2, MROWS_, 4 * H2_, H1_);

  // 6) LSTM layer 2 (16 WGs x 64 cols), writes Y2bf
  lstm_kernel<<<H2_ / LCOLS, 256, 0, stream>>>(xw2, U2bf, h2buf, Y2bf, tokens,
                                               ctr + T_, H2_, H2_ / LCOLS);

  // 7) logits = Y2 @ Wd + bd   [2048 x 50257]
  if (fast_dense) {
    dense_bf16_kernel<<<dim3(VP_ / 256, MROWS_ / 64), 256, 0, stream>>>(
        Y2bf, Wdbf, bd, out, MROWS_, V_, H2_);
  } else {
    dense_out_kernel<<<dim3((V_ + 255) / 256, MROWS_ / 64), 256, 0, stream>>>(
        Y2bf, Wd, bd, out, MROWS_, V_, H2_);
  }
}
